// SubjectBlock_20598663152244
// MI455X (gfx1250) — compile-verified, matrix operand order checked by reading, not run on hardware
//
#include <hip/hip_runtime.h>

// Problem constants (from reference)
constexpr int B_  = 128;
constexpr int C_  = 295;   // input channels (K of big GEMM)
constexpr int T_  = 2048;  // time (N of big GEMM)
constexpr int D1_ = 270;   // output channels (M)
constexpr int NS_ = 64;    // number of subjects
constexpr int KP_ = 320;   // padded K (10 steps of 32)
constexpr int MP_ = 320;   // padded M (multiple of 64)

typedef __attribute__((ext_vector_type(16))) __bf16 v16bf;
typedef __attribute__((ext_vector_type(8)))  float  v8f;

union Frag {
    uint4         q[2];   // two 16-byte halves
    unsigned int  u[8];
    v16bf         v;
};

__device__ inline unsigned short f2bf(float f) {
    union { float f; unsigned int u; } c; c.f = f;
    unsigned int x = c.u;
    unsigned int r = x + 0x7FFFu + ((x >> 16) & 1u);  // round-to-nearest-even
    return (unsigned short)(r >> 16);
}

#if __has_builtin(__builtin_amdgcn_cvt_pk_bf16_f32)
__device__ inline unsigned int pack_bf16(float f0, float f1) {
    typedef __attribute__((ext_vector_type(2))) __bf16 v2bf;
    union { v2bf v; unsigned int u; } c;
    c.v = __builtin_amdgcn_cvt_pk_bf16_f32(f0, f1);   // v_cvt_pk_bf16_f32
    return c.u;
}
#else
__device__ inline unsigned int pack_bf16(float f0, float f1) {
    return (unsigned int)f2bf(f0) | ((unsigned int)f2bf(f1) << 16);
}
#endif

// ---------------------------------------------------------------------------
// Pack kernels: branchless (clamped loads + select), write zero-padded bf16.
// W1t[c][k]  = W1[k][c]  (B-panel, transposed so B fragments are b128 loads)
// Wsp[s][m][k] = Ws[s][m][k]  (A-panel)
// ---------------------------------------------------------------------------
__global__ __launch_bounds__(256) void pack_w1_kernel(
    const float* __restrict__ W1, unsigned short* __restrict__ W1t)
{
    const int idx = blockIdx.x * 256 + threadIdx.x;   // over MP_*KP_
    const int c = idx / KP_;
    const int k = idx - c * KP_;
    if (c >= MP_) return;
    const int kc = (k < D1_) ? k : (D1_ - 1);
    const int cc = (c < C_)  ? c : (C_  - 1);
    float f = W1[kc * C_ + cc];
    f = (k < D1_ && c < C_) ? f : 0.f;
    W1t[(size_t)c * KP_ + k] = f2bf(f);
}

__global__ __launch_bounds__(256) void pack_ws_kernel(
    const float* __restrict__ Ws, unsigned short* __restrict__ Wsp)
{
    const int s   = blockIdx.y;
    const int idx = blockIdx.x * 256 + threadIdx.x;   // over MP_*KP_
    const int m = idx / KP_;
    const int k = idx - m * KP_;
    if (m >= MP_) return;
    const int mc = (m < D1_) ? m : (D1_ - 1);
    const int kc = (k < D1_) ? k : (D1_ - 1);
    float f = Ws[(size_t)s * D1_ * D1_ + mc * D1_ + kc];
    f = (m < D1_ && k < D1_) ? f : 0.f;
    Wsp[((size_t)s * MP_ + m) * KP_ + k] = f2bf(f);
}

// ---------------------------------------------------------------------------
// Phase A: W2[b] = Wsp[subj[b]] @ W1t^T  (bf16, zero-padded to [MP_ x KP_]).
// One wave per 16x16 tile; all loads are unguarded b128 in fragment layout.
// Zero padding in the packed inputs makes pad rows/cols of W2 zero naturally.
// ---------------------------------------------------------------------------
__global__ __launch_bounds__(32) void fuse_w2_kernel(
    const unsigned short* __restrict__ Wsp, const unsigned short* __restrict__ W1t,
    const int* __restrict__ sidx, unsigned short* __restrict__ W2)
{
    const int lane = threadIdx.x;
    const int n0 = blockIdx.x * 16;       // column tile (c)
    const int m0 = blockIdx.y * 16;       // row tile (e)
    const int b  = blockIdx.z;

    const int row = lane & 15;
    const int kb  = (lane >> 4) * 8;      // lanes 16-31 handle K+8
    const unsigned short* A0 = Wsp + ((size_t)sidx[b] * MP_ + m0 + row) * KP_;
    const unsigned short* B0 = W1t + (size_t)(n0 + row) * KP_;

    v8f acc = {};
    for (int ks = 0; ks < 9; ++ks) {      // K = 270 -> 288 (padded region is zero)
        const int k0 = ks * 32;
        Frag a, bm;
        a.q[0]  = *(const uint4*)(A0 + k0 + kb);
        a.q[1]  = *(const uint4*)(A0 + k0 + kb + 16);
        bm.q[0] = *(const uint4*)(B0 + k0 + kb);
        bm.q[1] = *(const uint4*)(B0 + k0 + kb + 16);
        acc = __builtin_amdgcn_wmma_f32_16x16x32_bf16(
                  false, a.v, false, bm.v, (short)0, acc, false, false);
    }

    const int half8 = (lane >> 4) * 8;
    const int cc = n0 + (lane & 15);
    #pragma unroll
    for (int r = 0; r < 8; ++r) {
        const int mm = m0 + r + half8;
        W2[((size_t)b * MP_ + mm) * KP_ + cc] = f2bf(acc[r]);  // pad already zero
    }
}

// ---------------------------------------------------------------------------
// bs[b] = Ws[subj[b]] @ b1  (f32, padded to MP_)
// ---------------------------------------------------------------------------
__global__ __launch_bounds__(MP_) void bias_fuse_kernel(
    const float* __restrict__ Ws, const float* __restrict__ b1,
    const int* __restrict__ sidx, float* __restrict__ bs)
{
    const int b = blockIdx.x;
    const int e = threadIdx.x;
    float sum = 0.f;
    if (e < D1_) {
        const float* Wrow = Ws + (size_t)sidx[b] * D1_ * D1_ + (size_t)e * D1_;
        for (int d = 0; d < D1_; ++d) sum += Wrow[d] * b1[d];
    }
    bs[b * MP_ + e] = sum;
}

// ---------------------------------------------------------------------------
// Phase B: out[b] = W2[b] @ X[b] + bs[b]
// Block = 256 threads (8 waves). Block tile: 64(M) x 128(N), one batch.
// A panel (bf16, no conversion needed) comes in via the Tensor Data Mover
// (tensor_load_to_lds + s_wait_tensorcnt); X tile is loaded coalesced,
// converted f32->bf16 in registers, staged in LDS transposed so each lane's
// B fragment is two ds_load_b128.
// ---------------------------------------------------------------------------
#if __has_builtin(__builtin_amdgcn_tensor_load_to_lds) && \
    __has_builtin(__builtin_amdgcn_s_wait_tensorcnt)
#define USE_TDM 1
typedef __attribute__((ext_vector_type(4))) unsigned int v4u;
typedef __attribute__((ext_vector_type(8))) int          v8i_;
typedef __attribute__((ext_vector_type(4))) int          v4i_;
#else
#define USE_TDM 0
#endif

__global__ __launch_bounds__(256) void subject_gemm_kernel(
    const float* __restrict__ X, const unsigned short* __restrict__ W2,
    const float* __restrict__ bs, float* __restrict__ out)
{
    constexpr int LROW = 40;                                 // LDS row stride pad
    __shared__ __align__(16) unsigned short Bt[128 * LROW];  // 10 KB  (X tile, bf16)
    __shared__ __align__(16) unsigned short At[64 * 32];     // 4 KB   (W2 tile)

    const int tid  = threadIdx.x;
    const int lane = tid & 31;
    const int w    = tid >> 5;
    const int b    = blockIdx.z;
    const int mb0  = blockIdx.y * 64;
    const int nb0  = blockIdx.x * 128;
    const int mt   = w & 3;                       // M-tile within block
    const int nh   = w >> 2;                      // N half (0/1)

    const int mrow  = lane & 15;
    const int kb    = (lane >> 4) * 8;
    const int half8 = (lane >> 4) * 8;

    // Accumulators initialized with fused bias bs[m]
    float bias[8];
    const float* bsb = bs + b * MP_ + mb0 + mt * 16;
    #pragma unroll
    for (int r = 0; r < 8; ++r) bias[r] = bsb[r + half8];
    v8f acc[4];
    #pragma unroll
    for (int nt = 0; nt < 4; ++nt)
        #pragma unroll
        for (int r = 0; r < 8; ++r) acc[nt][r] = bias[r];

    // Staging mapping: thread -> (column sn, K-half kq)
    const int sn = tid & 127;
    const int kq = tid >> 7;
    const float* Xcol = X + (size_t)b * C_ * T_ + nb0 + sn;

    const unsigned short* W2panel = W2 + ((size_t)b * MP_ + mb0) * KP_;

#if USE_TDM
    const unsigned int lds_at = (unsigned int)(unsigned long long)&At[0];
#endif

    for (int ks = 0; ks < 10; ++ks) {
        const int k0 = ks * 32;

        // ---- load X strip (coalesced, branchless clamp+select), pack bf16
        unsigned int pk[8];
        #pragma unroll
        for (int j = 0; j < 16; j += 2) {
            const int k  = k0 + kq * 16 + j;
            const int c0 = (k     < C_) ? k     : (C_ - 1);
            const int c1 = (k + 1 < C_) ? k + 1 : (C_ - 1);
            float f0 = Xcol[(size_t)c0 * T_];
            float f1 = Xcol[(size_t)c1 * T_];
            f0 = (k     < C_) ? f0 : 0.f;
            f1 = (k + 1 < C_) ? f1 : 0.f;
            pk[j >> 1] = pack_bf16(f0, f1);
        }
        if (ks + 1 < 10)
            __builtin_prefetch(Xcol + (size_t)(k0 + 32) * T_, 0, 3);

        __syncthreads();   // previous iteration done reading LDS

#if USE_TDM
        if (w == 0) {
            // Tensor DMA: 2-D tile, 32 (K, contiguous) x 64 (M rows), bf16,
            // row stride KP_, from padded W2 panel -> At (rows packed, 64 B each).
            const unsigned long long ga =
                (unsigned long long)(const void*)(W2panel + k0);
            v4u g0; v8i_ g1; v4i_ gz; v8i_ gz8;
            g0[0] = 1u;                                   // count=1, user mode
            g0[1] = lds_at;                               // lds_addr
            g0[2] = (unsigned int)ga;                     // global_addr[31:0]
            g0[3] = (unsigned int)((ga >> 32) & 0x1FFFFFFu) | (2u << 30); // addr[56:32] | type=2
            g1[0] = (int)(1u << 16);                      // data_size=1 (2 bytes)
            g1[1] = (int)((unsigned)KP_ << 16);           // tensor_dim0[15:0] in [31:16]
            g1[2] = (int)((unsigned)MP_ << 16);           // dim0 hi=0 | tensor_dim1[15:0]
            g1[3] = (int)(32u << 16);                     // dim1 hi=0 | tile_dim0=32
            g1[4] = 64;                                   // tile_dim1=64, tile_dim2=0
            g1[5] = KP_;                                  // tensor_dim0_stride (elements)
            g1[6] = 0;
            g1[7] = 0;
            gz[0] = 0; gz[1] = 0; gz[2] = 0; gz[3] = 0;
            gz8[0] = 0; gz8[1] = 0; gz8[2] = 0; gz8[3] = 0;
            gz8[4] = 0; gz8[5] = 0; gz8[6] = 0; gz8[7] = 0;
            __builtin_amdgcn_tensor_load_to_lds(g0, g1, gz, gz, gz8, 0);
        }
#endif
        // ---- stage B tile (transposed [n][k], bf16)
        uint4* dst = (uint4*)&Bt[sn * LROW + kq * 16];
        uint4 q0; q0.x = pk[0]; q0.y = pk[1]; q0.z = pk[2]; q0.w = pk[3];
        uint4 q1; q1.x = pk[4]; q1.y = pk[5]; q1.z = pk[6]; q1.w = pk[7];
        dst[0] = q0;
        dst[1] = q1;
#if !USE_TDM
        // Fallback: wave-parallel A staging via plain loads
        if (w < 2) {
            const int r64 = w * 32 + lane;                // 0..63 rows
            const uint4 av = *(const uint4*)(W2panel + (size_t)r64 * KP_ + k0);
            *(uint4*)&At[r64 * 32] = av;
            const uint4 av2 = *(const uint4*)(W2panel + (size_t)r64 * KP_ + k0 + 8);
            *(uint4*)&At[r64 * 32 + 8] = av2;
            const uint4 av3 = *(const uint4*)(W2panel + (size_t)r64 * KP_ + k0 + 16);
            *(uint4*)&At[r64 * 32 + 16] = av3;
            const uint4 av4 = *(const uint4*)(W2panel + (size_t)r64 * KP_ + k0 + 24);
            *(uint4*)&At[r64 * 32 + 24] = av4;
        }
#else
        if (w == 0) __builtin_amdgcn_s_wait_tensorcnt(0); // A tile landed in LDS
#endif
        __syncthreads();   // tiles visible to all waves

        // ---- A fragment: two ds_load_b128 from At
        Frag a;
        const unsigned short* asrc = &At[(mt * 16 + mrow) * 32 + kb];
        a.q[0] = *(const uint4*)asrc;
        a.q[1] = *(const uint4*)(asrc + 16);

        // ---- 4 N-tiles: B fragment from LDS + WMMA
        #pragma unroll
        for (int nt = 0; nt < 4; ++nt) {
            Frag bf;
            const unsigned short* bsrc = &Bt[(nh * 64 + nt * 16 + mrow) * LROW + kb];
            bf.q[0] = *(const uint4*)bsrc;
            bf.q[1] = *(const uint4*)(bsrc + 16);
            acc[nt] = __builtin_amdgcn_wmma_f32_16x16x32_bf16(
                          false, a.v, false, bf.v, (short)0, acc[nt], false, false);
        }
    }

    // ---- store D fragments (guard M pad rows)
    #pragma unroll
    for (int nt = 0; nt < 4; ++nt) {
        const int col = nb0 + nh * 64 + nt * 16 + (lane & 15);
        #pragma unroll
        for (int r = 0; r < 8; ++r) {
            const int mm = mb0 + mt * 16 + r + half8;
            if (mm < D1_)
                out[((size_t)b * D1_ + mm) * T_ + col] = acc[nt][r];
        }
    }
}

// ---------------------------------------------------------------------------
extern "C" void kernel_launch(void* const* d_in, const int* in_sizes, int n_in,
                              void* d_out, int out_size, void* d_ws, size_t ws_size,
                              hipStream_t stream)
{
    (void)in_sizes; (void)n_in; (void)out_size; (void)ws_size;
    const float* X    = (const float*)d_in[0];
    const int*   sidx = (const int*)  d_in[1];
    const float* W1   = (const float*)d_in[2];
    const float* b1   = (const float*)d_in[3];
    const float* Ws   = (const float*)d_in[4];
    float* out = (float*)d_out;

    // Workspace layout
    char* ws = (char*)d_ws;
    unsigned short* W2  = (unsigned short*)ws;                       // 128*320*320*2 = 26.2 MB
    ws += (size_t)B_ * MP_ * KP_ * 2;
    float* bsub = (float*)ws;                                        // 128*320*4 = 160 KB
    ws += (size_t)B_ * MP_ * 4;
    unsigned short* W1t = (unsigned short*)ws;                       // 320*320*2 = 200 KB
    ws += (size_t)MP_ * KP_ * 2;
    unsigned short* Wsp = (unsigned short*)ws;                       // 64*320*320*2 = 13.1 MB

    const int panel = MP_ * KP_ / 256;   // 400 blocks per padded panel

    // Pack padded bf16 operands (branchless)
    pack_w1_kernel<<<dim3(panel), 256, 0, stream>>>(W1, W1t);
    pack_ws_kernel<<<dim3(panel, NS_), 256, 0, stream>>>(Ws, Wsp);

    // Phase A: fuse weights + bias
    fuse_w2_kernel<<<dim3(KP_ / 16, MP_ / 16, B_), 32, 0, stream>>>(Wsp, W1t, sidx, W2);
    bias_fuse_kernel<<<dim3(B_), MP_, 0, stream>>>(Ws, b1, sidx, bsub);

    // Phase B: big memory-bound GEMM over X
    subject_gemm_kernel<<<dim3(T_ / 128, MP_ / 64, B_), 256, 0, stream>>>(X, W2, bsub, out);
}
